// MultiHeadAttention_65678639890569
// MI455X (gfx1250) — 8-bit matrix operand order repaired and checked by reading, compiled, not run on hardware
//
#include <hip/hip_runtime.h>
#include <hip/hip_bf16.h>
#include <stdint.h>


typedef float v8f __attribute__((ext_vector_type(8)));
typedef float v2f __attribute__((ext_vector_type(2)));
typedef int   v8i __attribute__((ext_vector_type(8)));
typedef int   vsi4 __attribute__((vector_size(16)));

#define E_DIM 256
#define N_DIM 512
#define T_DIM 4

#if defined(__has_builtin)
#if __has_builtin(__builtin_amdgcn_global_load_async_to_lds_b128)
#define HAVE_ASYNC_LDS 1
#endif
#endif
#ifndef HAVE_ASYNC_LDS
#define HAVE_ASYNC_LDS 0
#endif

static __device__ __forceinline__ void wait_asynccnt0() {
#if defined(__has_builtin) && __has_builtin(__builtin_amdgcn_s_wait_asynccnt)
  __builtin_amdgcn_s_wait_asynccnt(0);
#else
  asm volatile("s_wait_asynccnt 0x0" ::: "memory");
#endif
}

__global__ __launch_bounds__(128) void proj_gemm_kernel(
    const float* __restrict__ X, const float* __restrict__ W,
    const float* __restrict__ bias, float* __restrict__ Y) {
  const int lane = threadIdx.x & 31;
  const int wave = threadIdx.x >> 5;
  const int m0   = blockIdx.x * 16;
  const int f0   = wave * 64;
  const int rc   = lane & 15;
  const int half = lane >> 4;

  const float* xptr = X + (size_t)(m0 + rc) * E_DIM + half * 2;
  const float* wptr = W + (size_t)(f0 + rc) * E_DIM + half * 2;
  v8f acc0 = {}, acc1 = {}, acc2 = {}, acc3 = {};
#pragma unroll 4
  for (int k = 0; k < E_DIM; k += 4) {
    const v2f a  = *(const v2f*)(xptr + k);
    const v2f b0 = *(const v2f*)(wptr + k);
    const v2f b1 = *(const v2f*)(wptr + 16 * E_DIM + k);
    const v2f b2 = *(const v2f*)(wptr + 32 * E_DIM + k);
    const v2f b3 = *(const v2f*)(wptr + 48 * E_DIM + k);
    acc0 = __builtin_amdgcn_wmma_f32_16x16x4_f32(false, a, false, b0, (short)0,
                                                 acc0, false, false);
    acc1 = __builtin_amdgcn_wmma_f32_16x16x4_f32(false, a, false, b1, (short)0,
                                                 acc1, false, false);
    acc2 = __builtin_amdgcn_wmma_f32_16x16x4_f32(false, a, false, b2, (short)0,
                                                 acc2, false, false);
    acc3 = __builtin_amdgcn_wmma_f32_16x16x4_f32(false, a, false, b3, (short)0,
                                                 acc3, false, false);
  }
  const float bb0 = bias[f0 + rc];
  const float bb1 = bias[f0 + 16 + rc];
  const float bb2 = bias[f0 + 32 + rc];
  const float bb3 = bias[f0 + 48 + rc];
  float* yp = Y + (size_t)(m0 + half * 8) * E_DIM + f0 + rc;
#pragma unroll
  for (int r = 0; r < 8; ++r) {
    yp[(size_t)r * E_DIM + 0]  = acc0[r] + bb0;
    yp[(size_t)r * E_DIM + 16] = acc1[r] + bb1;
    yp[(size_t)r * E_DIM + 32] = acc2[r] + bb2;
    yp[(size_t)r * E_DIM + 48] = acc3[r] + bb3;
  }
}

__global__ __launch_bounds__(256) void lif_kernel(
    const float* __restrict__ Y, unsigned char* __restrict__ Su8,
    float* __restrict__ Sf32) {
  const int NE  = N_DIM * E_DIM;
  const int idx = blockIdx.x * blockDim.x + threadIdx.x;
  const int b   = idx / NE;
  const int rem = idx - b * NE;
  float mem = 0.0f;
#pragma unroll
  for (int t = 0; t < T_DIM; ++t) {
    const size_t off = (size_t)(b * T_DIM + t) * NE + rem;
    mem = mem * 0.5f + Y[off];
    const float sp = (mem > 1.0f) ? 1.0f : 0.0f;
    if (Su8)  Su8[off]  = (unsigned char)sp;
    if (Sf32) Sf32[off] = sp;
    mem = (1.0f - sp) * mem;
  }
}

__global__ __launch_bounds__(128) void attn_kernel(
    const unsigned char* __restrict__ Sq, const unsigned char* __restrict__ Sk,
    const float* __restrict__ Sv, float* __restrict__ O) {
  __shared__ float Esm[16 * N_DIM];
  __shared__ float Vsm[128 * 32];
  __shared__ float red[128];
  __shared__ float rowstat[16];

  const int tid  = threadIdx.x;
  const int lane = tid & 31;
  const int wave = tid >> 5;
  const int rb   = blockIdx.x;
  const int h    = blockIdx.y;
  const int bt   = blockIdx.z;
  const int rc   = lane & 15;
  const int half = lane >> 4;

  const unsigned char* qb =
      Sq + ((size_t)(bt * N_DIM + rb * 16 + rc)) * E_DIM + h * 32 + half * 16;
  v8i Afrag = { *(const int*)(qb),      *(const int*)(qb + 4),
                *(const int*)(qb + 8),  *(const int*)(qb + 12),
                0, 0, 0, 0 };

  for (int ct = wave; ct < 32; ct += 4) {
    const unsigned char* kb =
        Sk + ((size_t)(bt * N_DIM + ct * 16 + rc)) * E_DIM + h * 32 + half * 16;
    const int4 k4 = *(const int4*)kb;
    v8i Bfrag = { k4.x, k4.y, k4.z, k4.w, 0, 0, 0, 0 };
    v8i Ci = {};
    v8i Ei = __builtin_amdgcn_wmma_i32_16x16x64_iu8(
        false, Afrag, false, Bfrag, Ci, false, false);
#pragma unroll
    for (int r = 0; r < 8; ++r)
      Esm[(r + half * 8) * N_DIM + ct * 16 + rc] = (float)Ei[r];
  }
  __syncthreads();

  const int srow = tid & 15;
  const int sbas = (tid >> 4) * 64;
  float lmax = -1e30f;
  for (int j = 0; j < 64; ++j)
    lmax = fmaxf(lmax, Esm[srow * N_DIM + sbas + j]);
  red[tid] = lmax;
  __syncthreads();
  if (tid < 16) {
    float m = red[tid];
    for (int s = 1; s < 8; ++s) m = fmaxf(m, red[tid + s * 16]);
    rowstat[tid] = m;
  }
  __syncthreads();
  const float rmax = rowstat[srow];
  float lsum = 0.0f;
  for (int j = 0; j < 64; ++j) {
    const int ii = srow * N_DIM + sbas + j;
    const float p = __expf((Esm[ii] - rmax) * 0.0625f);
    Esm[ii] = p;
    lsum += p;
  }
  red[tid] = lsum;
  __syncthreads();
  if (tid < 16) {
    float s = 0.0f;
    for (int sg = 0; sg < 8; ++sg) s += red[tid + sg * 16];
    rowstat[tid] = 1.0f / s;
  }
  __syncthreads();
  const float rinv = rowstat[srow];
  for (int j = 0; j < 64; ++j) Esm[srow * N_DIM + sbas + j] *= rinv;
  __syncthreads();

  const int c0 = wave * 16;
  v8f acc = {};
  for (int c = 0; c < 4; ++c) {
    for (int i = tid; i < 1024; i += 128) {
      const int l  = i >> 3;
      const int dc = (i & 7) << 2;
      const float* gsrc =
          Sv + ((size_t)(bt * N_DIM + c * 128 + l)) * E_DIM + h * 32 + dc;
#if HAVE_ASYNC_LDS
      __builtin_amdgcn_global_load_async_to_lds_b128(
          (__attribute__((address_space(1))) vsi4*)gsrc,
          (__attribute__((address_space(3))) vsi4*)&Vsm[(l << 5) + dc],
          0, 0);
#else
      *(float4*)&Vsm[(l << 5) + dc] = *(const float4*)gsrc;
#endif
    }
#if HAVE_ASYNC_LDS
    wait_asynccnt0();
#endif
    __syncthreads();
    if (wave < 2) {
#pragma unroll 4
      for (int k = 0; k < 128; k += 4) {
        const int kk = k + half * 2;
        v2f a = { Esm[rc * N_DIM + c * 128 + kk],
                  Esm[rc * N_DIM + c * 128 + kk + 1] };
        v2f b = { Vsm[kk * 32 + c0 + rc], Vsm[(kk + 1) * 32 + c0 + rc] };
        acc = __builtin_amdgcn_wmma_f32_16x16x4_f32(
            false, a, false, b, (short)0, acc, false, false);
      }
    }
    __syncthreads();
  }
  if (wave < 2) {
    float* ob =
        O + ((size_t)(bt * N_DIM + rb * 16 + half * 8)) * E_DIM + h * 32 + c0 + rc;
#pragma unroll
    for (int r = 0; r < 8; ++r) ob[(size_t)r * E_DIM] = acc[r];
  }
}

extern "C" void kernel_launch(void* const* d_in, const int* in_sizes, int n_in,
                              void* d_out, int out_size, void* d_ws,
                              size_t ws_size, hipStream_t stream) {
  const float* Xq = (const float*)d_in[0];
  const float* Xk = (const float*)d_in[1];
  const float* Xv = (const float*)d_in[2];
  const float* Wq = (const float*)d_in[3];
  const float* bq = (const float*)d_in[4];
  const float* Wk = (const float*)d_in[5];
  const float* bk = (const float*)d_in[6];
  const float* Wv = (const float*)d_in[7];
  const float* bv = (const float*)d_in[8];
  const float* Wo = (const float*)d_in[9];
  const float* bo = (const float*)d_in[10];
  float* out = (float*)d_out;

  char* ws = (char*)d_ws;
  float*         Y  = (float*)(ws);
  unsigned char* Sq = (unsigned char*)(ws + (size_t)16 * 1024 * 1024);
  unsigned char* Sk = (unsigned char*)(ws + (size_t)20 * 1024 * 1024);
  float*         Sv = (float*)(ws + (size_t)24 * 1024 * 1024);
  float*         O  = (float*)(ws + (size_t)40 * 1024 * 1024);

  const dim3 gG(1024),      bG(128);
  const dim3 gL(4096),      bL(256);
  const dim3 gA(32, 8, 32), bA(128);

  proj_gemm_kernel<<<gG, bG, 0, stream>>>(Xq, Wq, bq, Y);
  lif_kernel<<<gL, bL, 0, stream>>>(Y, Sq, nullptr);
  proj_gemm_kernel<<<gG, bG, 0, stream>>>(Xk, Wk, bk, Y);
  lif_kernel<<<gL, bL, 0, stream>>>(Y, Sk, nullptr);
  proj_gemm_kernel<<<gG, bG, 0, stream>>>(Xv, Wv, bv, Y);
  lif_kernel<<<gL, bL, 0, stream>>>(Y, nullptr, Sv);
  attn_kernel<<<gA, bA, 0, stream>>>(Sq, Sk, Sv, O);
  proj_gemm_kernel<<<gG, bG, 0, stream>>>(O, Wo, bo, Y);
  lif_kernel<<<gL, bL, 0, stream>>>(Y, nullptr, out);
}
